// HybridMemory_35184372089415
// MI455X (gfx1250) — compile-verified, matrix operand order checked by reading, not run on hardware
//
#include <hip/hip_runtime.h>

// ---------------------------------------------------------------------------
// Problem constants (from the reference)
// ---------------------------------------------------------------------------
#define DD      4096        // feature dim
#define NL      18048       // bank rows (num labeled)
#define NROWS   512         // proposals in batch
#define INVT    20.0f       // 1 / TEMP
#define SHIFT   20.0f       // logits are in [-20,20]; exp(logit-20) is safe
#define CT      4           // column tiles (16 cols each) per wave

typedef __attribute__((ext_vector_type(16))) __bf16 v16bf;
typedef __attribute__((ext_vector_type(8)))  float  v8f;

union BF16x16 { v16bf v; unsigned int u[8]; };

__device__ __forceinline__ unsigned short f2bf(float f) {
  unsigned int u = __float_as_uint(f);
  u += 0x7FFFu + ((u >> 16) & 1u);          // round-to-nearest-even
  return (unsigned short)(u >> 16);
}

// ---------------------------------------------------------------------------
// 0) zero the accumulators used with atomics
// ---------------------------------------------------------------------------
__global__ void init_kernel(float* __restrict__ rowsum, float* __restrict__ scal) {
  int t = blockIdx.x * blockDim.x + threadIdx.x;
  if (t < NROWS) rowsum[t] = 0.0f;
  if (t == NROWS)     scal[0] = 0.0f;   // cos-sum
  if (t == NROWS + 1) scal[1] = 0.0f;   // kl-sum
}

// ---------------------------------------------------------------------------
// 1) fp32 -> packed bf16 streaming conversion (2 elems per uint)
// ---------------------------------------------------------------------------
__global__ void cvt_bf16_kernel(const float* __restrict__ src,
                                unsigned int* __restrict__ dst, int npairs) {
  int i = blockIdx.x * blockDim.x + threadIdx.x;
  int stride = gridDim.x * blockDim.x;
  for (; i < npairs; i += stride) {
    float2 f = ((const float2*)src)[i];
    dst[i] = (unsigned int)f2bf(f.x) | ((unsigned int)f2bf(f.y) << 16);
  }
}

// ---------------------------------------------------------------------------
// fragment loaders
// ---------------------------------------------------------------------------
__device__ __forceinline__ void load_afrag(BF16x16& a,
                                           const unsigned short* __restrict__ ldsA,
                                           int ln, int hl, int k0) {
  uint4 a0 = *(const uint4*)&ldsA[ln * 1024 + k0 + hl * 8];
  uint4 a1 = *(const uint4*)&ldsA[ln * 1024 + k0 + 16 + hl * 8];
  a.u[0] = a0.x; a.u[1] = a0.y; a.u[2] = a0.z; a.u[3] = a0.w;
  a.u[4] = a1.x; a.u[5] = a1.y; a.u[6] = a1.z; a.u[7] = a1.w;
}

__device__ __forceinline__ void load_bfrag(BF16x16& b,
                                           const unsigned short* __restrict__ p,
                                           int kG, int hl) {
  uint4 b0 = *(const uint4*)&p[kG + hl * 8];
  uint4 b1 = *(const uint4*)&p[kG + 16 + hl * 8];
  b.u[0] = b0.x; b.u[1] = b0.y; b.u[2] = b0.z; b.u[3] = b0.w;
  b.u[4] = b1.x; b.u[5] = b1.y; b.u[6] = b1.z; b.u[7] = b1.w;
}

// ---------------------------------------------------------------------------
// 2) bf16 WMMA GEMM  C = A @ B^T  (A: [M,4096] bf16 rows, B: [L,4096] bf16 rows)
//    WG = 8 waves = 16 rows x 512 cols; each wave holds 4 C tiles (16x64).
//    K staged via LDS in 1024-chunks; fragments double-buffered so B global
//    loads for step k+1 overlap the 4 WMMAs of step k.
//    mode==1: fused softmax-CE denominator: rowsum[r] += sum_c exp(20*sim-20)
//    mode==0: write raw fp32 tile to out (ldOut)
// ---------------------------------------------------------------------------
__global__ void gemm_bf16_nt_kernel(const unsigned short* __restrict__ A,
                                    const unsigned short* __restrict__ B,
                                    int L, int mode,
                                    float* __restrict__ out, int ldOut,
                                    float* __restrict__ rowsum) {
  __shared__ unsigned short ldsA[16 * 1024];   // 32 KB A K-chunk
  __shared__ float rowAcc[16];

  const int tid  = threadIdx.x;
  const int wave = tid >> 5;
  const int lane = tid & 31;
  const int hl   = lane >> 4;          // lane half (0: lanes 0-15, 1: 16-31)
  const int ln   = lane & 15;
  const int rowBase = blockIdx.x * 16;
  const int colWave = blockIdx.y * 512 + wave * 64;

  if (tid < 16) rowAcc[tid] = 0.0f;

  v8f c[CT];
  const unsigned short* __restrict__ Bptr[CT];
  bool valid[CT];
#pragma unroll
  for (int t = 0; t < CT; ++t) {
    c[t] = v8f{};
    int col  = colWave + t * 16 + ln;
    valid[t] = col < L;
    Bptr[t]  = B + (size_t)(valid[t] ? col : (L - 1)) * DD;
  }

  for (int kc = 0; kc < 4; ++kc) {
    const int kcBase = kc * 1024;
    __syncthreads();
    // cooperative copy of A[rowBase..+16, kcBase..+1024] (bf16) into LDS
    {
      const unsigned int* src =
          (const unsigned int*)(A + (size_t)rowBase * DD + kcBase);
      unsigned int* dst = (unsigned int*)ldsA;
      for (int i = tid; i < 16 * 512; i += 256) {
        int r = i >> 9, kk = i & 511;                 // 512 uints per row-chunk
        dst[r * 512 + kk] = src[(size_t)r * (DD / 2) + kk];
      }
    }
    __syncthreads();

    // -------- double-buffered K loop: 32 steps of K=32 --------
    BF16x16 a0, a1, b0[CT], b1[CT];
    load_afrag(a0, ldsA, ln, hl, 0);
#pragma unroll
    for (int t = 0; t < CT; ++t) load_bfrag(b0[t], Bptr[t], kcBase, hl);

#pragma unroll 2
    for (int ks = 0; ks < 32; ks += 2) {
      // issue loads for step ks+1 into buffer 1, then MMA buffer 0
      if (ks + 1 < 32) {
        load_afrag(a1, ldsA, ln, hl, (ks + 1) * 32);
#pragma unroll
        for (int t = 0; t < CT; ++t)
          load_bfrag(b1[t], Bptr[t], kcBase + (ks + 1) * 32, hl);
      }
#pragma unroll
      for (int t = 0; t < CT; ++t)
        c[t] = __builtin_amdgcn_wmma_f32_16x16x32_bf16(
                   false, a0.v, false, b0[t].v, (short)0, c[t], false, false);

      // issue loads for step ks+2 into buffer 0, then MMA buffer 1
      if (ks + 2 < 32) {
        load_afrag(a0, ldsA, ln, hl, (ks + 2) * 32);
#pragma unroll
        for (int t = 0; t < CT; ++t)
          load_bfrag(b0[t], Bptr[t], kcBase + (ks + 2) * 32, hl);
      }
#pragma unroll
      for (int t = 0; t < CT; ++t)
        c[t] = __builtin_amdgcn_wmma_f32_16x16x32_bf16(
                   false, a1.v, false, b1[t].v, (short)0, c[t], false, false);
    }
  }

  // C/D layout: VGPR i, lanes 0-15 -> row i; lanes 16-31 -> row i+8; col = ln
  if (mode == 0) {
#pragma unroll
    for (int t = 0; t < CT; ++t) {
      if (!valid[t]) continue;
#pragma unroll
      for (int i = 0; i < 8; ++i) {
        int row = rowBase + i + hl * 8;
        out[(size_t)row * ldOut + colWave + t * 16 + ln] = c[t][i];
      }
    }
  } else {
#pragma unroll
    for (int t = 0; t < CT; ++t) {
#pragma unroll
      for (int i = 0; i < 8; ++i) {
        float e = valid[t] ? __expf(c[t][i] * INVT - SHIFT) : 0.0f;
        e += __shfl_xor(e, 1, 16);
        e += __shfl_xor(e, 2, 16);
        e += __shfl_xor(e, 4, 16);
        e += __shfl_xor(e, 8, 16);          // sum over the 16 cols of this tile
        if (ln == 0) atomicAdd(&rowAcc[hl * 8 + i], e);
      }
    }
    __syncthreads();
    if (tid < 16) atomicAdd(&rowsum[rowBase + tid], rowAcc[tid]);
  }
}

// ---------------------------------------------------------------------------
// 3) target logits (feat_cat[n] . bank[pids[n]]) * 20  and cosine sum
//    one wave per row
// ---------------------------------------------------------------------------
__global__ void aux_kernel(const float* __restrict__ feat_cat,
                           const float* __restrict__ bank,
                           const int*   __restrict__ pids,
                           const float* __restrict__ features,
                           const float* __restrict__ features1,
                           float* __restrict__ tgt,
                           float* __restrict__ cossum) {
  const int wave = threadIdx.x >> 5, lane = threadIdx.x & 31;
  const int n = blockIdx.x * 8 + wave;
  const float4* a = (const float4*)(feat_cat  + (size_t)n * DD);
  const float4* b = (const float4*)(bank      + (size_t)pids[n] * DD);
  const float4* f = (const float4*)(features  + (size_t)n * DD);
  const float4* g = (const float4*)(features1 + (size_t)n * DD);
  float accT = 0.0f, accC = 0.0f;
  for (int i = lane; i < DD / 4; i += 32) {
    float4 av = a[i], bv = b[i];
    accT += av.x * bv.x + av.y * bv.y + av.z * bv.z + av.w * bv.w;
    float4 fv = f[i], gv = g[i];
    accC += fv.x * gv.x + fv.y * gv.y + fv.z * gv.z + fv.w * gv.w;
  }
  for (int m = 16; m; m >>= 1) {
    accT += __shfl_xor(accT, m, 32);
    accC += __shfl_xor(accC, m, 32);
  }
  if (lane == 0) {
    tgt[n] = accT * INVT;
    atomicAdd(cossum, accC);
  }
}

// ---------------------------------------------------------------------------
// 4) symmetric KL between row softmaxes of simP / simQ (one block per row)
// ---------------------------------------------------------------------------
__global__ void kl_kernel(const float* __restrict__ simP,
                          const float* __restrict__ simQ,
                          float* __restrict__ klsum) {
  __shared__ float red[256];
  __shared__ float lse[2];
  const int row = blockIdx.x, tid = threadIdx.x;
  const float* p = simP + (size_t)row * NROWS;
  const float* q = simQ + (size_t)row * NROWS;

  float sp = 0.0f, sq = 0.0f;
  for (int j = tid; j < NROWS; j += 256) { sp += __expf(p[j]); sq += __expf(q[j]); }

  red[tid] = sp; __syncthreads();
  for (int s = 128; s; s >>= 1) { if (tid < s) red[tid] += red[tid + s]; __syncthreads(); }
  if (tid == 0) lse[0] = __logf(red[0]);
  __syncthreads();
  red[tid] = sq; __syncthreads();
  for (int s = 128; s; s >>= 1) { if (tid < s) red[tid] += red[tid + s]; __syncthreads(); }
  if (tid == 0) lse[1] = __logf(red[0]);
  __syncthreads();

  const float lp0 = lse[0], lq0 = lse[1];
  float acc = 0.0f;
  for (int j = tid; j < NROWS; j += 256) {
    float lp = p[j] - lp0, lq = q[j] - lq0;
    acc += __expf(lq) * (lq - lp) + __expf(lp) * (lp - lq);
  }
  red[tid] = acc; __syncthreads();
  for (int s = 128; s; s >>= 1) { if (tid < s) red[tid] += red[tid + s]; __syncthreads(); }
  if (tid == 0) atomicAdd(klsum, red[0]);
}

// ---------------------------------------------------------------------------
// 5) final scalar
// ---------------------------------------------------------------------------
__global__ void finalize_kernel(const float* __restrict__ rowsum,
                                const float* __restrict__ tgt,
                                const float* __restrict__ scal,
                                float* __restrict__ outv) {
  __shared__ float red[512];
  const int tid = threadIdx.x;
  // CE_n = logsumexp - target_logit  (shifted exp => +SHIFT)
  red[tid] = __logf(rowsum[tid]) + SHIFT - tgt[tid];
  __syncthreads();
  for (int s = 256; s; s >>= 1) { if (tid < s) red[tid] += red[tid + s]; __syncthreads(); }
  if (tid == 0) {
    float ce_mean = red[0] * (1.0f / (float)NROWS);
    float loss1   = __logf((float)NL);                 // CE over zero logits
    float losscos = 1.0f - scal[0] * (1.0f / (float)NROWS);
    outv[0] = 0.5f * (ce_mean + loss1) + losscos + scal[1];
  }
}

// ---------------------------------------------------------------------------
// launch
// ---------------------------------------------------------------------------
extern "C" void kernel_launch(void* const* d_in, const int* in_sizes, int n_in,
                              void* d_out, int out_size, void* d_ws, size_t ws_size,
                              hipStream_t stream) {
  (void)in_sizes; (void)n_in; (void)out_size; (void)ws_size;
  const float* feat_cat  = (const float*)d_in[0];
  const float* features1 = (const float*)d_in[1];
  const float* features  = (const float*)d_in[2];
  const int*   pids      = (const int*)  d_in[3];
  const float* bank      = (const float*)d_in[4];
  float* outv = (float*)d_out;

  // workspace carve-up (byte offsets, all >=256B aligned)
  char* ws = (char*)d_ws;
  float*          rowsum   = (float*)(ws + 0);                       //  2 KB
  float*          tgt      = (float*)(ws + 2048);                    //  2 KB
  float*          scal     = (float*)(ws + 4096);                    //  cos,kl
  float*          simP     = (float*)(ws + 8192);                    //  1 MB
  float*          simQ     = (float*)(ws + 8192 + 1048576);          //  1 MB
  unsigned short* bankBf   = (unsigned short*)(ws + 8192 + 2 * 1048576);
  unsigned short* fcatBf   = (unsigned short*)((char*)bankBf + (size_t)NL * DD * 2);
  unsigned short* featBf   = (unsigned short*)((char*)fcatBf + (size_t)NROWS * DD * 2);
  unsigned short* feat1Bf  = (unsigned short*)((char*)featBf + (size_t)NROWS * DD * 2);

  init_kernel<<<3, 256, 0, stream>>>(rowsum, scal);

  // fp32 -> bf16 (bank dominates: 296 MB read + 148 MB write, pure streaming)
  cvt_bf16_kernel<<<4096, 256, 0, stream>>>(bank,      (unsigned int*)bankBf,
                                            NL * DD / 2);
  cvt_bf16_kernel<<<512, 256, 0, stream>>>(feat_cat,  (unsigned int*)fcatBf,
                                            NROWS * DD / 2);
  cvt_bf16_kernel<<<512, 256, 0, stream>>>(features,  (unsigned int*)featBf,
                                            NROWS * DD / 2);
  cvt_bf16_kernel<<<512, 256, 0, stream>>>(features1, (unsigned int*)feat1Bf,
                                            NROWS * DD / 2);

  // big GEMM with fused exp-rowsum: grid.x (row tiles) fastest so the 32 WGs
  // sharing one bank strip are dispatch-adjacent -> L2 captures the reuse.
  // 36 column strips of 512 (last strip partially masked: 36*512 >= 18048)
  gemm_bf16_nt_kernel<<<dim3(32, 36), 256, 0, stream>>>(
      fcatBf, bankBf, NL, /*mode=*/1, simP /*unused*/, NROWS, rowsum);

  // self-similarity matrices (raw fp32 tiles); 512 cols == one strip
  gemm_bf16_nt_kernel<<<dim3(32, 1), 256, 0, stream>>>(
      featBf, featBf, NROWS, /*mode=*/0, simP, NROWS, rowsum);
  gemm_bf16_nt_kernel<<<dim3(32, 1), 256, 0, stream>>>(
      feat1Bf, feat1Bf, NROWS, /*mode=*/0, simQ, NROWS, rowsum);

  aux_kernel<<<NROWS / 8, 256, 0, stream>>>(feat_cat, bank, pids,
                                            features, features1, tgt, scal);

  kl_kernel<<<NROWS, 256, 0, stream>>>(simP, simQ, scal + 1);

  finalize_kernel<<<1, NROWS, 0, stream>>>(rowsum, tgt, scal, outv);
}
